// GENConvNet_22935125360680
// MI455X (gfx1250) — matrix operand order repaired and checked by reading, not compiled, not run on hardware
//
#include <hip/hip_runtime.h>

// ---------------- problem constants (match reference) ----------------
#define N_NODES 50000
#define N_EDGES 1600000
#define NBATCH  64
#define EMB_D   16
#define HID_D   192
#define OUT_F   128
#define DEMO_D  5
#define OUT_D   2
#define GEN_EPS 1e-7f

typedef __attribute__((ext_vector_type(16))) _Float16 v16h;
typedef __attribute__((ext_vector_type(8)))  float    v8f;

// ---------------- embedding gather: x[i, :] = emb[node_idx[i], :] ----------------
__global__ void gather_embed_kernel(const int* __restrict__ node_idx,
                                    const float* __restrict__ emb,
                                    float* __restrict__ x) {
  int t = blockIdx.x * blockDim.x + threadIdx.x;
  if (t >= N_NODES * EMB_D) return;
  int i = t >> 4, d = t & 15;
  x[t] = emb[(node_idx[i] << 4) + d];
}

// ---------------- agg := x (implements the "+ x_i" of GENConv) ----------------
__global__ void copy_f4_kernel(const float4* __restrict__ in,
                               float4* __restrict__ out, int n4) {
  int t = blockIdx.x * blockDim.x + threadIdx.x;
  if (t < n4) out[t] = in[t];
}

// ---------------- weights -> f16, pre-packed into WMMA B-fragment layout ------
// Whp layout: fragment-major [ct][kc][lane][h], 16 halves (32 B) contiguous per
// lane, so the GEMM loads each B fragment as one v16h (two global_load_b128,
// coalesced across the wave).
//   lane L (N = ct*16 + L%16), elem h -> K = kc*32 + h + 16*(L>=16)
__global__ void pack_w_kernel(const float* __restrict__ W, _Float16* __restrict__ Whp,
                              int Kreal, int Kpad, int Kout) {
  int t = blockIdx.x * blockDim.x + threadIdx.x;
  if (t >= Kpad * Kout) return;
  int h    = t & 15;
  int lane = (t >> 4) & 31;
  int rest = t >> 9;
  int nkc  = Kpad >> 5;
  int kc   = rest % nkc;
  int ct   = rest / nkc;
  int k    = (kc << 5) + h + ((lane >> 4) << 4);
  int col  = (ct << 4) + (lane & 15);
  Whp[t] = (k < Kreal) ? (_Float16)W[k * Kout + col] : (_Float16)0.0f;
}

// ---------------- fused relu(x[src])+eps scatter-add into agg[dst] ----------------
// One thread = one edge x 4 channels; D4 is compile-time so the edge/channel
// decomposition strength-reduces to mul-shift. Consecutive threads cover
// consecutive channels of one edge -> coalesced float4 gathers; f32 atomics
// resolve in L2 (x + agg working set ~77MB fits the 192MB L2).
template<int D4>
__global__ void scatter_relu_kernel(const float* __restrict__ x,
                                    const int* __restrict__ src,
                                    const int* __restrict__ dst,
                                    float* __restrict__ agg,
                                    long total) {
  long t = (long)blockIdx.x * blockDim.x + threadIdx.x;
  if (t >= total) return;
  int e  = (int)(t / D4);
  int c4 = (int)(t - (long)e * D4);
  int s = src[e];
  int d = dst[e];
  const float4 v = ((const float4*)x)[(size_t)s * D4 + c4];
  float4 m;
  m.x = fmaxf(v.x, 0.0f) + GEN_EPS;
  m.y = fmaxf(v.y, 0.0f) + GEN_EPS;
  m.z = fmaxf(v.z, 0.0f) + GEN_EPS;
  m.w = fmaxf(v.w, 0.0f) + GEN_EPS;
  float* p = agg + (((size_t)d * D4 + c4) << 2);
  atomicAdd(p + 0, m.x);
  atomicAdd(p + 1, m.y);
  atomicAdd(p + 2, m.z);
  atomicAdd(p + 3, m.w);
}

// ---------------- WMMA GEMM: Y[N,KOUT] = A[N,Kreal] @ W[KPAD,KOUT] + bias -----
// Block = 128 threads (4 wave32s) = 16 rows; wave w owns column tiles w, w+4...
// A tile staged in LDS as f16 (padded stride); B fragments loaded as v16h from
// the pre-packed layout. KPAD/KOUT compile-time -> fully unrolled K and ct
// loops, back-to-back v_wmma with pipelined ds/vmem loads.
// Fragment layouts (C/D per CDNA5 ISA 7.12.2):
//   A (16-bit 16x32, this code's k order, matching B packing): lane L (M=L%16), elem h -> K = h + 16*(L>=16)
//   C/D (f32 16x16):  vgpr r, lane L -> M = r + 8*(L>=16), N = L%16
template<int KPAD, int KOUT>
__global__ void __launch_bounds__(128)
gemm_wmma_kernel(const float* __restrict__ A, const _Float16* __restrict__ Whp,
                 const float* __restrict__ bias, float* __restrict__ Y, int Kreal) {
  constexpr int LDA = KPAD + 8;     // halves; padded to spread LDS banks
  constexpr int NKC = KPAD / 32;    // k-chunks of 32
  constexpr int NCT = KOUT / 16;    // 16-wide column tiles
  __shared__ _Float16 As[16 * LDA];

  const int m0  = blockIdx.x << 4;
  const int tid = threadIdx.x;

  // cooperative A-tile load + f32->f16 convert, zero-pad K in [Kreal, KPAD)
  for (int idx = tid; idx < 16 * KPAD; idx += 128) {
    int r = idx / KPAD;
    int k = idx - r * KPAD;
    float v = (k < Kreal) ? A[(size_t)(m0 + r) * Kreal + k] : 0.0f;
    As[r * LDA + k] = (_Float16)v;
  }
  __syncthreads();

  const int wave   = tid >> 5;        // wave32 on gfx1250
  const int lane   = tid & 31;
  const int laneHi = lane >> 4;       // 0 | 1
  const int ln     = lane & 15;
  const v16h* __restrict__ Bp = (const v16h*)Whp;

#pragma unroll
  for (int i = 0; i < NCT / 4; ++i) {
    const int ct   = wave + i * 4;
    const int col0 = ct << 4;
    v8f acc = {};
#pragma unroll
    for (int kc = 0; kc < NKC; ++kc) {
      const int k0 = kc << 5;
      v16h a;
#pragma unroll
      for (int h = 0; h < 16; ++h)
        a[h] = As[ln * LDA + k0 + h + (laneHi << 4)];
      const v16h b = Bp[(ct * NKC + kc) * 32 + lane];   // 32B contiguous/lane
      acc = __builtin_amdgcn_wmma_f32_16x16x32_f16(false, a, false, b,
                                                   (short)0, acc, false, false);
    }
    const float bcol = bias[col0 + ln];
#pragma unroll
    for (int r = 0; r < 8; ++r) {
      int row = m0 + r + (laneHi << 3);
      Y[(size_t)row * KOUT + col0 + ln] = acc[r] + bcol;
    }
  }
}

// ---------------- pooling ----------------
__global__ void zero_pool_kernel(float* __restrict__ gsum, float* __restrict__ cnt) {
  int t = blockIdx.x * blockDim.x + threadIdx.x;
  if (t < NBATCH * OUT_F) gsum[t] = 0.0f;
  else if (t < NBATCH * OUT_F + NBATCH) cnt[t - NBATCH * OUT_F] = 0.0f;
}

__global__ void count_kernel(const int* __restrict__ batch, float* __restrict__ cnt) {
  int i = blockIdx.x * blockDim.x + threadIdx.x;
  if (i < N_NODES) atomicAdd(&cnt[batch[i]], 1.0f);
}

__global__ void pool_sum_kernel(const float* __restrict__ x,
                                const int* __restrict__ batch,
                                float* __restrict__ gsum) {
  int t = blockIdx.x * blockDim.x + threadIdx.x;
  if (t >= N_NODES * OUT_F) return;
  int i = t >> 7, c = t & (OUT_F - 1);
  atomicAdd(&gsum[batch[i] * OUT_F + c], x[t]);
}

// ---------------- MLP head: one block per graph ----------------
__global__ void head_kernel(const float* __restrict__ gsum, const float* __restrict__ cnt,
                            const float* __restrict__ demo,
                            const float* __restrict__ Wc1, const float* __restrict__ bc1,
                            const float* __restrict__ Wc2, const float* __restrict__ bc2,
                            float* __restrict__ out) {
  __shared__ float feats[OUT_F + DEMO_D];   // 133
  __shared__ float h[64];
  const int b = blockIdx.x;
  const int j = threadIdx.x;                // 64 threads
  const float inv = 1.0f / fmaxf(cnt[b], 1.0f);
  for (int c = j; c < OUT_F; c += 64) feats[c] = gsum[b * OUT_F + c] * inv;
  if (j < DEMO_D) feats[OUT_F + j] = demo[b * DEMO_D + j];
  __syncthreads();
  float s = bc1[j];
  for (int c = 0; c < OUT_F + DEMO_D; ++c) s += feats[c] * Wc1[c * 64 + j];
  h[j] = fmaxf(s, 0.0f);
  __syncthreads();
  if (j < OUT_D) {
    float o = bc2[j];
    for (int c = 0; c < 64; ++c) o += h[c] * Wc2[c * OUT_D + j];
    out[b * OUT_D + j] = o;
  }
}

// ---------------- host orchestration ----------------
static inline size_t algn256(size_t x) { return (x + 255) & ~(size_t)255; }

extern "C" void kernel_launch(void* const* d_in, const int* in_sizes, int n_in,
                              void* d_out, int out_size, void* d_ws, size_t ws_size,
                              hipStream_t stream) {
  (void)in_sizes; (void)n_in; (void)out_size; (void)ws_size;

  const int*   node_idx = (const int*)d_in[0];
  const int*   eidx     = (const int*)d_in[1];   // [2, E]
  const int*   batch    = (const int*)d_in[2];
  const float* demo     = (const float*)d_in[3];
  const float* emb      = (const float*)d_in[4];
  const float* W[4]  = {(const float*)d_in[5], (const float*)d_in[7],
                        (const float*)d_in[9], (const float*)d_in[11]};
  const float* Bv[4] = {(const float*)d_in[6], (const float*)d_in[8],
                        (const float*)d_in[10], (const float*)d_in[12]};
  const float* Wc1 = (const float*)d_in[13];
  const float* bc1 = (const float*)d_in[14];
  const float* Wc2 = (const float*)d_in[15];
  const float* bc2 = (const float*)d_in[16];

  const int* src = eidx;              // edge_index[0]
  const int* dst = eidx + N_EDGES;    // edge_index[1]

  const int Kreal[4] = {EMB_D, HID_D, HID_D, HID_D};
  const int Kpad[4]  = {32,    HID_D, HID_D, HID_D};
  const int Kout[4]  = {HID_D, HID_D, HID_D, OUT_F};

  // workspace carve-up (~77 MB)
  char* ws = (char*)d_ws;
  size_t off = 0;
  float* bufX = (float*)(ws + off); off += algn256((size_t)N_NODES * HID_D * 4);
  float* bufA = (float*)(ws + off); off += algn256((size_t)N_NODES * HID_D * 4);
  _Float16* Whp[4];
  for (int l = 0; l < 4; ++l) {
    Whp[l] = (_Float16*)(ws + off);
    off += algn256((size_t)Kpad[l] * Kout[l] * 2);
  }
  float* gsum = (float*)(ws + off); off += algn256((size_t)NBATCH * OUT_F * 4);
  float* cnt  = (float*)(ws + off); off += algn256((size_t)NBATCH * 4);

  // pack weights to f16 fragment layout — cheap, deterministic, capturable
  for (int l = 0; l < 4; ++l) {
    int total = Kpad[l] * Kout[l];
    pack_w_kernel<<<(total + 255) / 256, 256, 0, stream>>>(W[l], Whp[l], Kreal[l], Kpad[l], Kout[l]);
  }

  gather_embed_kernel<<<(N_NODES * EMB_D + 255) / 256, 256, 0, stream>>>(node_idx, emb, bufX);

  for (int l = 0; l < 4; ++l) {
    const int Din = Kreal[l];
    const int n4  = N_NODES * Din / 4;
    copy_f4_kernel<<<(n4 + 255) / 256, 256, 0, stream>>>((const float4*)bufX, (float4*)bufA, n4);

    const long total = (long)N_EDGES * (Din / 4);
    const unsigned nb = (unsigned)((total + 255) / 256);
    if (Din == EMB_D)
      scatter_relu_kernel<EMB_D / 4><<<nb, 256, 0, stream>>>(bufX, src, dst, bufA, total);
    else
      scatter_relu_kernel<HID_D / 4><<<nb, 256, 0, stream>>>(bufX, src, dst, bufA, total);

    if (l == 0)
      gemm_wmma_kernel<32, HID_D><<<N_NODES / 16, 128, 0, stream>>>(bufA, Whp[l], Bv[l], bufX, Din);
    else if (l < 3)
      gemm_wmma_kernel<HID_D, HID_D><<<N_NODES / 16, 128, 0, stream>>>(bufA, Whp[l], Bv[l], bufX, Din);
    else
      gemm_wmma_kernel<HID_D, OUT_F><<<N_NODES / 16, 128, 0, stream>>>(bufA, Whp[l], Bv[l], bufX, Din);
  }

  zero_pool_kernel<<<(NBATCH * OUT_F + NBATCH + 255) / 256, 256, 0, stream>>>(gsum, cnt);
  count_kernel<<<(N_NODES + 255) / 256, 256, 0, stream>>>(batch, cnt);
  pool_sum_kernel<<<(N_NODES * OUT_F + 255) / 256, 256, 0, stream>>>(bufX, batch, gsum);
  head_kernel<<<NBATCH, 64, 0, stream>>>(gsum, cnt, demo, Wc1, bc1, Wc2, bc2, (float*)d_out);
}
